// MoireAttention_30940944400472
// MI455X (gfx1250) — matrix operand order repaired and checked by reading, not compiled, not run on hardware
//
#include <hip/hip_runtime.h>
#include <cmath>
#include <stdint.h>
#include <stddef.h>

// ---------------- problem constants ----------------
#define B_    2
#define T_    2048
#define C_    1024
#define H_    16
#define D_    64
#define TWO_C 2048

typedef __attribute__((ext_vector_type(16))) __bf16    v16bf;
typedef __attribute__((ext_vector_type(8)))  float     v8f;
typedef __attribute__((ext_vector_type(4)))  uint32_t  u32x4;
typedef __attribute__((ext_vector_type(8)))  uint32_t  u32x8;

union Frag {
    uint4 q[2];
    v16bf v;
};

__device__ __forceinline__ __bf16 tobf(float f) {
    union { float f; uint32_t u; } a; a.f = f;
    uint32_t r = a.u + 0x7FFFu + ((a.u >> 16) & 1u);
    uint16_t h = (uint16_t)(r >> 16);
    __bf16 o; __builtin_memcpy(&o, &h, sizeof(o));
    return o;
}

__device__ __forceinline__ v8f wmma_bf16(v16bf a, v16bf b, v8f c) {
    return __builtin_amdgcn_wmma_f32_16x16x32_bf16(
        false, a, false, b, (short)0, c, false, false);
}

__device__ __forceinline__ uint32_t lds_addr32(const void* p) {
    // generic LDS pointer: low 32 bits are the LDS offset (ISA 10.2 aperture rule)
    return (uint32_t)(uintptr_t)p;
}

// ---- CDNA5 Tensor Data Mover: 2D bf16 tile, global -> LDS (TENSORcnt-tracked) ----
// D# per CDNA5 ISA: group0 = {count/type/addrs}, group1 = dims.
__device__ __forceinline__ void tdm_load_tile_bf16(
    uint32_t lds_off, const void* gaddr,
    uint32_t tensor_d0, uint32_t tensor_d1, uint64_t stride0 /*elements*/,
    uint32_t tile_d0, uint32_t tile_d1)
{
    uint64_t ga = (uint64_t)(uintptr_t)gaddr;
    u32x4 g0;
    g0[0] = 1u;                                               // count=1, user desc
    g0[1] = lds_off;                                          // lds_addr
    g0[2] = (uint32_t)ga;                                     // global_addr[31:0]
    g0[3] = (uint32_t)((ga >> 32) & 0x01FFFFFFu) | (2u << 30);// addr[56:32] | type=2
    u32x8 g1;
    g1[0] = (1u << 16);                                       // mask=0, data_size=1 (2B)
    g1[1] = (tensor_d0 & 0xFFFFu) << 16;                      // tensor_dim0[15:0]
    g1[2] = (tensor_d0 >> 16) | ((tensor_d1 & 0xFFFFu) << 16);
    g1[3] = (tensor_d1 >> 16) | (tile_d0 << 16);              // tile_dim0
    g1[4] = tile_d1;                                          // tile_dim1 (tile_dim2=0)
    g1[5] = (uint32_t)stride0;                                // tensor_dim0_stride[31:0]
    g1[6] = (uint32_t)((stride0 >> 32) & 0xFFFFu);            // stride0[47:32], stride1=0
    g1[7] = 0u;
    asm volatile("tensor_load_to_lds %0, %1" :: "s"(g0), "s"(g1) : "memory");
}

// ---------------- f32 -> bf16 converts ----------------
__global__ __launch_bounds__(256) void cvt_f32_bf16_kernel(
    const float* __restrict__ src, __bf16* __restrict__ dst, int n) {
    int i = blockIdx.x * blockDim.x + threadIdx.x;
    if (i < n) dst[i] = tobf(src[i]);
}

// weights: [K][N] f32 -> [N][K] bf16 (pre-transposed so TDM can tile it directly)
__global__ __launch_bounds__(256) void cvt_tr_bf16_kernel(
    const float* __restrict__ src, __bf16* __restrict__ dst, int K, int N) {
    int i = blockIdx.x * blockDim.x + threadIdx.x;
    if (i >= K * N) return;
    int k = i / N, n = i - k * N;
    dst[(size_t)n * K + k] = tobf(src[i]);
}

// ---------------- tiled bf16 GEMM: C(f32)[M,N] = A[M,K] * Bt[N,K]^T ----------------
#define GBM 128
#define GBN 128
#define GBK 32

__global__ __launch_bounds__(256) void gemm_bf16_f32_kernel(
    const __bf16* __restrict__ A,   // [M][K] row-major
    const __bf16* __restrict__ Bt,  // [N][K] row-major (pre-transposed B)
    float* __restrict__ Cm,         // [M][N] row-major
    int M, int N, int K)
{
    __shared__ __align__(16) __bf16 sA[2][GBM][GBK];   // [m][k]
    __shared__ __align__(16) __bf16 sB[2][GBN][GBK];   // [n][k]

    const int tid  = threadIdx.x;
    const int wave = tid >> 5;
    const int lane = tid & 31;
    const int lm = lane & 15, g = lane >> 4;
    const int bm = blockIdx.x * GBM;
    const int bn = blockIdx.y * GBN;
    const int wm = (wave >> 1) * 32;   // 4 waves along M
    const int wn = (wave & 1) * 64;    // 2 waves along N

    v8f acc[2][4] = {};

    const int nIter = K / GBK;
    auto issue = [&](int it, int buf) {
        const __bf16* ag = &A [(size_t)bm * K + it * GBK];
        const __bf16* bg = &Bt[(size_t)bn * K + it * GBK];
        tdm_load_tile_bf16(lds_addr32(&sA[buf][0][0]), ag, GBK, GBM, (uint64_t)K, GBK, GBM);
        tdm_load_tile_bf16(lds_addr32(&sB[buf][0][0]), bg, GBK, GBN, (uint64_t)K, GBK, GBN);
    };

    if (wave == 0) issue(0, 0);                    // prologue: fill buffer 0

    for (int it = 0; it < nIter; ++it) {
        const int cur = it & 1;
        if (wave == 0) {
            if (it + 1 < nIter) {
                issue(it + 1, cur ^ 1);            // prefetch next tile via TDM
                __builtin_amdgcn_s_wait_tensorcnt(2);   // current tile (2 ops) done
            } else {
                __builtin_amdgcn_s_wait_tensorcnt(0);
            }
        }
        __syncthreads();                            // tile visible to all waves

        Frag af[2], bfr[4];
#pragma unroll
        for (int i = 0; i < 2; ++i) {
            const __bf16* p = &sA[cur][wm + i * 16 + lm][0];
            af[i].q[0] = *(const uint4*)(p + 16 * g);       // K = 16g..16g+7
            af[i].q[1] = *(const uint4*)(p + 16 * g + 8);   // K = 16g+8..16g+15
        }
#pragma unroll
        for (int j = 0; j < 4; ++j) {
            const __bf16* p = &sB[cur][wn + j * 16 + lm][0];
            bfr[j].q[0] = *(const uint4*)(p + 16 * g);      // K = 16g..16g+15
            bfr[j].q[1] = *(const uint4*)(p + 16 * g + 8);
        }
#pragma unroll
        for (int i = 0; i < 2; ++i)
#pragma unroll
            for (int j = 0; j < 4; ++j)
                acc[i][j] = wmma_bf16(af[i].v, bfr[j].v, acc[i][j]);

        __syncthreads();                            // reads done before next overwrite
    }

#pragma unroll
    for (int i = 0; i < 2; ++i)
#pragma unroll
        for (int j = 0; j < 4; ++j)
#pragma unroll
            for (int r = 0; r < 8; ++r) {
                int row = bm + wm + i * 16 + r + 8 * g;
                int col = bn + wn + j * 16 + lm;
                Cm[(size_t)row * N + col] = acc[i][j][r];
            }
}

// ---------------- pack kernels ----------------
__global__ __launch_bounds__(256) void qk_pack_kernel(
    const float* __restrict__ raw, __bf16* __restrict__ pack, float scale, int n) {
    int i = blockIdx.x * blockDim.x + threadIdx.x;
    if (i >= n) return;
    int d = i & (D_ - 1);
    int h = (i >> 6) & (H_ - 1);
    int t = (i >> 10) & (T_ - 1);
    int b = i >> 21;                       // T_*C_ == 2^21
    size_t row = (size_t)(b * T_ + t) * TWO_C;
    float amp = raw[row + h * D_ + d];
    float ph  = raw[row + C_ + h * D_ + d];
    float a  = (amp > 20.0f) ? amp : log1pf(__expf(amp));   // softplus
    float pc = a * __cosf(ph) * scale;
    float ps = a * __sinf(ph) * scale;
    size_t o = ((size_t)(b * H_ + h) * T_ + t) * 128 + d;
    pack[o]      = tobf(pc);
    pack[o + D_] = tobf(ps);
}

// V packed TRANSPOSED: [B][H][D][T] so TDM delivers [d][key] tiles with no LDS scatter
__global__ __launch_bounds__(256) void v_pack_kernel(
    const float* __restrict__ vraw, __bf16* __restrict__ vt, int n) {
    int i = blockIdx.x * blockDim.x + threadIdx.x;
    if (i >= n) return;
    int d = i & (D_ - 1);
    int h = (i >> 6) & (H_ - 1);
    int t = (i >> 10) & (T_ - 1);
    int b = i >> 21;
    vt[((size_t)(b * H_ + h) * D_ + d) * T_ + t] = tobf(vraw[i]);
}

__global__ __launch_bounds__(256) void gate_init_kernel(
    const float* __restrict__ theta, float* __restrict__ gc, float* __restrict__ gs) {
    int i = blockIdx.x * blockDim.x + threadIdx.x;
    if (i >= H_ * T_) return;
    int h = i / T_, t = i - h * T_;
    float c = theta[h] * ((float)t * (1.0f / 8.0f));   // GAMMA = 8
    gc[i] = __cosf(c);
    gs[i] = __sinf(c);
}

// ---------------- causal flash attention with moiré gate ----------------
#define QT 128   // queries per workgroup (8 waves * 16 rows)
#define KT 32    // keys per step

__global__ __launch_bounds__(256) void flash_attn_kernel(
    const __bf16* __restrict__ qpack,   // [B*H][T][128]
    const __bf16* __restrict__ kpack,   // [B*H][T][128]
    const __bf16* __restrict__ vtpk,    // [B*H][D][T]  (pre-transposed V)
    const float*  __restrict__ gcos,    // [H][T]
    const float*  __restrict__ gsin,    // [H][T]
    __bf16* __restrict__ aout)          // [B][T][C] bf16
{
    __shared__ __align__(16) __bf16 sK[2][KT][128];    // keys row-major (== B [n][k])
    __shared__ __align__(16) __bf16 sV[2][D_][KT];     // V [d][key]
    __shared__ __align__(16) __bf16 sP[8][16][KT];     // per-wave P staging (C -> A layout)

    const int tid = threadIdx.x, wave = tid >> 5, lane = tid & 31;
    const int lm = lane & 15, g = lane >> 4;
    const int bh = blockIdx.y, b = bh / H_, h = bh % H_;
    const int qblk  = blockIdx.x * QT;
    const int qrow0 = qblk + wave * 16;

    const __bf16* qb = qpack + (size_t)bh * T_ * 128;
    const __bf16* kb = kpack + (size_t)bh * T_ * 128;
    const __bf16* vt = vtpk  + (size_t)bh * D_ * T_;
    const float* gcp = gcos + (size_t)h * T_;
    const float* gsp = gsin + (size_t)h * T_;

    // Q fragments (A-matrix layout): rows qrow0..+15, dims 0..127 in 4 chunks of 32
    Frag qf[4];
    {
        const __bf16* qrow = qb + (size_t)(qrow0 + lm) * 128;
#pragma unroll
        for (int f = 0; f < 4; ++f) {
            qf[f].q[0] = *(const uint4*)(qrow + 32 * f + 16 * g);
            qf[f].q[1] = *(const uint4*)(qrow + 32 * f + 16 * g + 8);
        }
    }
    float gq_c[8], gq_s[8];
#pragma unroll
    for (int r = 0; r < 8; ++r) {
        int q = qrow0 + r + 8 * g;
        gq_c[r] = gcp[q];
        gq_s[r] = gsp[q];
    }

    float mrow[8], lrow[8];
#pragma unroll
    for (int r = 0; r < 8; ++r) { mrow[r] = -INFINITY; lrow[r] = 0.0f; }
    v8f oacc[4] = {};

    const int nIter = (qblk + QT) / KT;    // causal upper bound for this block
    auto issueKV = [&](int it, int buf) {
        const int k0 = it * KT;
        // K tile: 32 keys x 128 dims from [T][128]
        tdm_load_tile_bf16(lds_addr32(&sK[buf][0][0]), kb + (size_t)k0 * 128,
                           128, KT, 128ull, 128, KT);
        // V tile: 64 d-rows x 32 keys from [D][T] (strided rows, TDM compacts into LDS)
        tdm_load_tile_bf16(lds_addr32(&sV[buf][0][0]), vt + k0,
                           KT, D_, (uint64_t)T_, KT, D_);
    };

    if (wave == 0) issueKV(0, 0);

    for (int it = 0; it < nIter; ++it) {
        const int cur = it & 1;
        const int k0  = it * KT;
        if (wave == 0) {
            if (it + 1 < nIter) {
                issueKV(it + 1, cur ^ 1);
                __builtin_amdgcn_s_wait_tensorcnt(2);
            } else {
                __builtin_amdgcn_s_wait_tensorcnt(0);
            }
        }
        __syncthreads();

        if (k0 <= qrow0 + 15) {            // tile not fully above the diagonal
            // S = Q * K^T over dim 128 : two 16x16 tiles (keys 0-15 / 16-31)
            v8f s0 = {}, s1 = {};
#pragma unroll
            for (int f = 0; f < 4; ++f) {
                Frag b0, b1;
                const __bf16* p0 = &sK[cur][lm][32 * f + 16 * g];
                const __bf16* p1 = &sK[cur][16 + lm][32 * f + 16 * g];
                b0.q[0] = *(const uint4*)p0;       b0.q[1] = *(const uint4*)(p0 + 8);
                b1.q[0] = *(const uint4*)p1;       b1.q[1] = *(const uint4*)(p1 + 8);
                s0 = wmma_bf16(qf[f].v, b0.v, s0);
                s1 = wmma_bf16(qf[f].v, b1.v, s1);
            }
            const float gk_c0 = gcp[k0 + lm],      gk_s0 = gsp[k0 + lm];
            const float gk_c1 = gcp[k0 + 16 + lm], gk_s1 = gsp[k0 + 16 + lm];

#pragma unroll
            for (int r = 0; r < 8; ++r) {
                const int qr = qrow0 + r + 8 * g;
                // moiré gate: cos(theta*(ck-cq)) = cq_c*ck_c + cq_s*ck_s
                float sc0 = s0[r] * (gq_c[r] * gk_c0 + gq_s[r] * gk_s0);
                float sc1 = s1[r] * (gq_c[r] * gk_c1 + gq_s[r] * gk_s1);
                if (k0 + lm > qr)      sc0 = -INFINITY;   // causal mask
                if (k0 + 16 + lm > qr) sc1 = -INFINITY;
                float mx = fmaxf(sc0, sc1);
#pragma unroll
                for (int off = 8; off >= 1; off >>= 1)
                    mx = fmaxf(mx, __shfl_xor(mx, off, 16));
                float newm = fmaxf(mrow[r], mx);
                float alpha = (mrow[r] == -INFINITY) ? 0.0f : __expf(mrow[r] - newm);
                float p0 = (sc0 == -INFINITY) ? 0.0f : __expf(sc0 - newm);
                float p1 = (sc1 == -INFINITY) ? 0.0f : __expf(sc1 - newm);
                float ps = p0 + p1;
#pragma unroll
                for (int off = 8; off >= 1; off >>= 1)
                    ps += __shfl_xor(ps, off, 16);
                lrow[r] = lrow[r] * alpha + ps;
                mrow[r] = newm;
#pragma unroll
                for (int j = 0; j < 4; ++j) oacc[j][r] *= alpha;
                sP[wave][r + 8 * g][lm]      = tobf(p0);
                sP[wave][r + 8 * g][16 + lm] = tobf(p1);
            }
            // reload P as A fragment (16x32), O += P * V
            Frag pf;
            const __bf16* pp = &sP[wave][lm][0];
            pf.q[0] = *(const uint4*)(pp + 16 * g);
            pf.q[1] = *(const uint4*)(pp + 16 * g + 8);
#pragma unroll
            for (int j = 0; j < 4; ++j) {
                Frag bv;
                const __bf16* vp = &sV[cur][j * 16 + lm][16 * g];
                bv.q[0] = *(const uint4*)vp;
                bv.q[1] = *(const uint4*)(vp + 8);
                oacc[j] = wmma_bf16(pf.v, bv.v, oacc[j]);
            }
        }
        __syncthreads();
    }

    // epilogue: normalize and scatter to [B][T][C] bf16 for the Wo GEMM
#pragma unroll
    for (int j = 0; j < 4; ++j)
#pragma unroll
        for (int r = 0; r < 8; ++r) {
            int qr = qrow0 + r + 8 * g;
            float val = oacc[j][r] * (1.0f / lrow[r]);
            aout[((size_t)b * T_ + qr) * C_ + h * D_ + j * 16 + lm] = tobf(val);
        }
}

// ---------------- host driver ----------------
extern "C" void kernel_launch(void* const* d_in, const int* in_sizes, int n_in,
                              void* d_out, int out_size, void* d_ws, size_t ws_size,
                              hipStream_t stream) {
    (void)in_sizes; (void)n_in; (void)out_size; (void)ws_size;
    const float* x  = (const float*)d_in[0];
    const float* Wq = (const float*)d_in[1];
    const float* Wk = (const float*)d_in[2];
    const float* Wv = (const float*)d_in[3];
    const float* Wo = (const float*)d_in[4];
    const float* th = (const float*)d_in[5];
    float* out = (float*)d_out;

    char* ws = (char*)d_ws;
    size_t off = 0;
    auto alloc = [&](size_t bytes) -> void* {
        void* p = ws + off;
        off = (off + bytes + 255) & ~(size_t)255;
        return p;
    };

    const int M  = B_ * T_;            // 4096
    const int NX = B_ * T_ * C_;       // 4194304

    __bf16* xb   = (__bf16*)alloc((size_t)NX * 2);
    __bf16* Wqt  = (__bf16*)alloc((size_t)C_ * TWO_C * 2);   // [2C][C]
    __bf16* Wkt  = (__bf16*)alloc((size_t)C_ * TWO_C * 2);   // [2C][C]
    __bf16* Wvt  = (__bf16*)alloc((size_t)C_ * C_ * 2);      // [C][C]
    __bf16* Wot  = (__bf16*)alloc((size_t)C_ * C_ * 2);      // [C][C]
    float*  qraw = (float*) alloc((size_t)M * TWO_C * 4);
    float*  kraw = (float*) alloc((size_t)M * TWO_C * 4);
    float*  vraw = (float*) alloc((size_t)NX * 4);
    __bf16* qpk  = (__bf16*)alloc((size_t)B_ * H_ * T_ * 128 * 2);
    __bf16* kpk  = (__bf16*)alloc((size_t)B_ * H_ * T_ * 128 * 2);
    __bf16* vtpk = (__bf16*)alloc((size_t)B_ * H_ * D_ * T_ * 2);
    float*  gc   = (float*) alloc((size_t)H_ * T_ * 4);
    float*  gs   = (float*) alloc((size_t)H_ * T_ * 4);
    __bf16* aob  = (__bf16*)alloc((size_t)NX * 2);

    const int TPB = 256;
    // 1) convert x to bf16; weights to bf16 transposed [N][K]
    cvt_f32_bf16_kernel<<<(NX + TPB - 1) / TPB, TPB, 0, stream>>>(x, xb, NX);
    cvt_tr_bf16_kernel<<<(C_ * TWO_C + TPB - 1) / TPB, TPB, 0, stream>>>(Wq, Wqt, C_, TWO_C);
    cvt_tr_bf16_kernel<<<(C_ * TWO_C + TPB - 1) / TPB, TPB, 0, stream>>>(Wk, Wkt, C_, TWO_C);
    cvt_tr_bf16_kernel<<<(C_ * C_ + TPB - 1) / TPB, TPB, 0, stream>>>(Wv, Wvt, C_, C_);
    cvt_tr_bf16_kernel<<<(C_ * C_ + TPB - 1) / TPB, TPB, 0, stream>>>(Wo, Wot, C_, C_);

    // 2) projection GEMMs (WMMA + TDM double-buffered tiles)
    dim3 gq(M / GBM, TWO_C / GBN);
    gemm_bf16_f32_kernel<<<gq, TPB, 0, stream>>>(xb, Wqt, qraw, M, TWO_C, C_);
    gemm_bf16_f32_kernel<<<gq, TPB, 0, stream>>>(xb, Wkt, kraw, M, TWO_C, C_);
    dim3 gv(M / GBM, C_ / GBN);
    gemm_bf16_f32_kernel<<<gv, TPB, 0, stream>>>(xb, Wvt, vraw, M, C_, C_);

    // 3) packing + gate tables
    qk_pack_kernel<<<(NX + TPB - 1) / TPB, TPB, 0, stream>>>(qraw, qpk, 0.125f, NX); // 1/sqrt(D)
    qk_pack_kernel<<<(NX + TPB - 1) / TPB, TPB, 0, stream>>>(kraw, kpk, 1.0f, NX);
    v_pack_kernel<<<(NX + TPB - 1) / TPB, TPB, 0, stream>>>(vraw, vtpk, NX);
    gate_init_kernel<<<(H_ * T_ + TPB - 1) / TPB, TPB, 0, stream>>>(th, gc, gs);

    // 4) flash attention with moiré gate (TDM-streamed K/V tiles)
    dim3 gf(T_ / QT, B_ * H_);
    flash_attn_kernel<<<gf, TPB, 0, stream>>>(qpk, kpk, vtpk, gc, gs, aob);

    // 5) output projection -> f32 d_out
    dim3 go(M / GBM, C_ / GBN);
    gemm_bf16_f32_kernel<<<go, TPB, 0, stream>>>(aob, Wot, out, M, C_, C_);
}
